// _TinyAttention_41137196761170
// MI455X (gfx1250) — matrix operand order repaired and checked by reading, not compiled, not run on hardware
//
#include <hip/hip_runtime.h>
#include <stdint.h>

// ---------------------------------------------------------------------------
// TinyAttention for MI455X (gfx1250, wave32, WMMA)
//   B=2, S=2048, E=2048, H=16, D=128.  ~206 GFLOP vs ~134 MB of HBM traffic
//   -> heavily compute bound; run every GEMM through V_WMMA_F32_16X16X32_BF16
//   with f32 accumulation, f32 softmax statistics.
// ---------------------------------------------------------------------------

#define HIDDEN 2048
#define NHEADS 16
#define HDIM   128
#define BATCH  2
#define SEQ    2048
#define TOKENS (BATCH * SEQ) /* 4096 */

typedef __attribute__((ext_vector_type(16))) __bf16          bf16x16;
typedef __attribute__((ext_vector_type(8)))  float           v8f;
typedef __attribute__((ext_vector_type(16))) unsigned short  u16x16;
typedef __attribute__((ext_vector_type(8)))  unsigned short  u16x8;
typedef int v4i_gcc __attribute__((vector_size(16)));  // matches builtin proto

static __device__ __forceinline__ unsigned short f32_to_bf16(float f) {
  unsigned int u = __builtin_bit_cast(unsigned int, f);
  u += 0x7fffu + ((u >> 16) & 1u);            // round-to-nearest-even
  return (unsigned short)(u >> 16);
}

static __device__ __forceinline__ bf16x16 pack_frag(u16x8 a, u16x8 b) {
  u16x16 t = __builtin_shufflevector(a, b, 0, 1, 2, 3, 4, 5, 6, 7,
                                     8, 9, 10, 11, 12, 13, 14, 15);
  return __builtin_bit_cast(bf16x16, t);
}

// A fragment (16x32 MxK, bf16): lanes 0-15 hold row M=lane with K {0..15},
// lanes 16-31 hold row M=lane-16 with K {16..31} (same order as B fragment).
static __device__ __forceinline__ bf16x16
load_frag_a(const unsigned short* base, int row0, int ld, int k0) {
  int lane = threadIdx.x & 31;
  int m = lane & 15, hi = lane >> 4;
  const unsigned short* p = base + (size_t)(row0 + m) * ld + (k0 + hi * 16);
  u16x8 c0 = *(const u16x8*)p;
  u16x8 c1 = *(const u16x8*)(p + 8);
  return pack_frag(c0, c1);
}

// B fragment (32x16 KxN, bf16): lane holds column N=lane&15, K=(lane>>4)*16..+15.
// `base` stores B^T row-major (row n == column n of B), stride ld, so the
// 16 K-elements per lane are one contiguous 32B load.
static __device__ __forceinline__ bf16x16
load_frag_b(const unsigned short* base, int col0, int ld, int k0) {
  int lane = threadIdx.x & 31;
  int n = lane & 15, hi = lane >> 4;
  const unsigned short* p = base + (size_t)(col0 + n) * ld + (k0 + hi * 16);
  return __builtin_bit_cast(bf16x16, *(const u16x16*)p);
}

static __device__ __forceinline__ v8f wmma_bf16(bf16x16 a, bf16x16 b, v8f c) {
  return __builtin_amdgcn_wmma_f32_16x16x32_bf16(
      /*neg_a=*/false, a, /*neg_b=*/false, b,
      /*c_mod=*/(short)0, c, /*reuse_a=*/false, /*reuse_b=*/false);
}

// --- async global->LDS staging (CDNA5 GLOBAL_LOAD_ASYNC_TO_LDS_B128, tracked
// --- by ASYNCcnt), with a load+ds_store fallback if the builtin is absent.
#if __has_builtin(__builtin_amdgcn_global_load_async_to_lds_b128)
#define HAVE_ASYNC_LDS 1
#endif

static __device__ __forceinline__ void stage16(const unsigned short* g,
                                               unsigned short* l) {
#ifdef HAVE_ASYNC_LDS
  __builtin_amdgcn_global_load_async_to_lds_b128(
      (__attribute__((address_space(1))) v4i_gcc*)g,
      (__attribute__((address_space(3))) v4i_gcc*)l,
      0, 0);
#else
  *(u16x8*)l = *(const u16x8*)g;
#endif
}

static __device__ __forceinline__ void stage_wait() {
#ifdef HAVE_ASYNC_LDS
#if __has_builtin(__builtin_amdgcn_s_wait_asynccnt)
  __builtin_amdgcn_s_wait_asynccnt(0);
#else
  asm volatile("s_wait_asynccnt 0x0" ::: "memory");
#endif
#endif
}

// ---------------------------------------------------------------------------
// fp32 -> bf16 conversion (pure bandwidth, ~9us total at 23.3 TB/s)
// ---------------------------------------------------------------------------
__global__ void cvt_f32_bf16_kernel(const float* __restrict__ in,
                                    unsigned short* __restrict__ out, int n) {
  int i = blockIdx.x * blockDim.x + threadIdx.x;
  if (i < n) out[i] = f32_to_bf16(in[i]);
}

// ---------------------------------------------------------------------------
// QKV projection: [4096,2048] x Wq/k/v^T.  Block tile 128x256, 8 waves in a
// 2(M) x 4(N) grid; each wave owns a 64x64 register tile: 4 A-frags + 4
// B-frags -> 16 WMMAs per 32-wide k-step (0.5 fragment loads per WMMA).
// Q,K stored [B*H,S,D]; V stored transposed [B*H,D,S] so attention's P@V
// B-operand is K-contiguous.
// ---------------------------------------------------------------------------
__global__ __launch_bounds__(256) void qkv_gemm_kernel(
    const unsigned short* __restrict__ X,
    const unsigned short* __restrict__ Wq,
    const unsigned short* __restrict__ Wk,
    const unsigned short* __restrict__ Wv,
    unsigned short* __restrict__ Q,
    unsigned short* __restrict__ K,
    unsigned short* __restrict__ Vt) {
  int w = threadIdx.x >> 5;
  int lane = threadIdx.x & 31;
  int m0 = blockIdx.x * 128 + (w >> 2) * 64;
  int n0 = blockIdx.y * 256 + (w & 3) * 64;
  int which = blockIdx.z;
  const unsigned short* W = (which == 0) ? Wq : (which == 1) ? Wk : Wv;

  v8f acc[4][4] = {};
  for (int k0 = 0; k0 < HIDDEN; k0 += 32) {
    bf16x16 a[4], b[4];
#pragma unroll
    for (int i = 0; i < 4; ++i) a[i] = load_frag_a(X, m0 + i * 16, HIDDEN, k0);
#pragma unroll
    for (int j = 0; j < 4; ++j) b[j] = load_frag_b(W, n0 + j * 16, HIDDEN, k0);
#pragma unroll
    for (int i = 0; i < 4; ++i)
#pragma unroll
      for (int j = 0; j < 4; ++j) acc[i][j] = wmma_bf16(a[i], b[j], acc[i][j]);
  }

  int n = lane & 15, hi = lane >> 4;
#pragma unroll
  for (int i = 0; i < 4; ++i) {
#pragma unroll
    for (int j = 0; j < 4; ++j) {
#pragma unroll
      for (int r = 0; r < 8; ++r) {
        int t = m0 + i * 16 + r + 8 * hi;  // token index
        int f = n0 + j * 16 + n;           // feature index
        int b_ = t >> 11, s = t & (SEQ - 1);
        int h = f >> 7, d = f & (HDIM - 1);
        unsigned short val = f32_to_bf16(acc[i][j][r]);
        size_t bh = (size_t)(b_ * NHEADS + h);
        if (which == 2)
          Vt[(bh * HDIM + d) * SEQ + s] = val;
        else if (which == 1)
          K[(bh * SEQ + s) * HDIM + d] = val;
        else
          Q[(bh * SEQ + s) * HDIM + d] = val;
      }
    }
  }
}

// ---------------------------------------------------------------------------
// Causal flash attention.  Grid (S/128, B*H).  8 waves; wave w owns queries
// [q0blk + w*16, +16).  kv loop is uniform across the block (nkv = bx+1, the
// diagonal block handled by masking) so the 128-key K and V tiles are staged
// ONCE per block into LDS (async global->LDS when available) instead of 8x
// per wave from global.  Online softmax per accumulator register row (rows
// live in 16-lane halves -> shfl_xor reductions).  P restaged through a
// per-wave LDS slice to convert C-layout f32 -> A-layout bf16.
// LDS: K 32KB + V 32KB + P 32KB = 96KB (of 320KB/WGP), dynamic.
// ---------------------------------------------------------------------------
__global__ __launch_bounds__(256) void attn_kernel(
    const unsigned short* __restrict__ Q,
    const unsigned short* __restrict__ K,
    const unsigned short* __restrict__ Vt,
    unsigned short* __restrict__ Oat) { // [TOKENS, HIDDEN] == [B,S,H*D] bf16
  extern __shared__ unsigned short smem[];
  unsigned short* ldsK = smem;            // [128 keys][128 d]
  unsigned short* ldsV = smem + 16384;    // [128 d][128 local keys]
  unsigned short* ldsP = smem + 32768;    // 8 x [16][128]

  const float C = 0.08838834764831845f * 1.4426950408889634f; // D^-0.5*log2(e)
  int w = threadIdx.x >> 5, lane = threadIdx.x & 31;
  int n = lane & 15, hi = lane >> 4;
  int bh = blockIdx.y;
  int h = bh & (NHEADS - 1), bb = bh >> 4;
  int q0 = blockIdx.x * 128 + w * 16;

  const unsigned short* Qh = Q + (size_t)bh * SEQ * HDIM;
  const unsigned short* Kh = K + (size_t)bh * SEQ * HDIM;
  const unsigned short* Vh = Vt + (size_t)bh * HDIM * SEQ;
  unsigned short* myP = ldsP + w * (16 * 128);

  bf16x16 aq[4];
#pragma unroll
  for (int kb = 0; kb < 4; ++kb) aq[kb] = load_frag_a(Qh, q0, HDIM, kb * 32);

  v8f o[8] = {};
  float mrow[8], lrow[8];
#pragma unroll
  for (int r = 0; r < 8; ++r) { mrow[r] = -3.0e38f; lrow[r] = 0.0f; }

  int nkv = blockIdx.x + 1;
  for (int it = 0; it < nkv; ++it) {
    int kv0 = it * 128;
    __syncthreads(); // WAR: everyone done reading previous K/V tiles
    {
      int tid = threadIdx.x;
      // K block: keys kv0..kv0+127 are one contiguous 32KB region of [S,D].
      const unsigned short* Kg = Kh + (size_t)kv0 * HDIM;
#pragma unroll
      for (int i = 0; i < 8; ++i) {
        int c = tid + i * 256;                 // 16B chunk id (2048 total)
        stage16(Kg + c * 8, ldsK + c * 8);
      }
      // V block: [D,S] -> 128 rows of 256B, row stride S.
#pragma unroll
      for (int i = 0; i < 8; ++i) {
        int c = tid + i * 256;
        int row = c >> 4, off = (c & 15) * 8;
        stage16(Vh + (size_t)row * SEQ + kv0 + off, ldsV + row * 128 + off);
      }
      stage_wait();
    }
    __syncthreads(); // RAW: staged tiles visible to all waves

    // scores = Q_tile @ K_block^T  (16 x 128), B-frags from LDS
    v8f sc[8] = {};
#pragma unroll
    for (int kb = 0; kb < 4; ++kb) {
#pragma unroll
      for (int nt = 0; nt < 8; ++nt) {
        bf16x16 bk = load_frag_b(ldsK, nt * 16, HDIM, kb * 32);
        sc[nt] = wmma_bf16(aq[kb], bk, sc[nt]);
      }
    }
    // online softmax per register row (row m = r + 8*hi)
#pragma unroll
    for (int r = 0; r < 8; ++r) {
      int qrow = q0 + r + 8 * hi;
      float rmax = -3.0e38f;
#pragma unroll
      for (int nt = 0; nt < 8; ++nt) {
        float s = sc[nt][r];
        if (kv0 + nt * 16 + n > qrow) s = -3.0e38f; // causal mask
        sc[nt][r] = s;
        rmax = fmaxf(rmax, s);
      }
#pragma unroll
      for (int off = 1; off < 16; off <<= 1)
        rmax = fmaxf(rmax, __shfl_xor(rmax, off, 32));
      float mnew = fmaxf(mrow[r], rmax);
      float alpha = exp2f((mrow[r] - mnew) * C);
      float rsum = 0.0f;
#pragma unroll
      for (int nt = 0; nt < 8; ++nt) {
        float p = exp2f((sc[nt][r] - mnew) * C);
        rsum += p;
        myP[(r + 8 * hi) * 128 + nt * 16 + n] = f32_to_bf16(p);
      }
#pragma unroll
      for (int off = 1; off < 16; off <<= 1)
        rsum += __shfl_xor(rsum, off, 32);
      lrow[r] = lrow[r] * alpha + rsum;
      mrow[r] = mnew;
#pragma unroll
      for (int dt = 0; dt < 8; ++dt) o[dt][r] *= alpha;
    }
    // P writes are cross-lane within the wave; LDS ops are in-order per wave,
    // so a dscnt wait suffices (no block barrier needed for the private slice).
    asm volatile("s_wait_dscnt 0x0" ::: "memory");
    // O += P @ V_block   (16 x 128) x (128 x 128), both operands from LDS
#pragma unroll
    for (int kb = 0; kb < 4; ++kb) {
      bf16x16 ap = load_frag_a(myP, 0, 128, kb * 32);
#pragma unroll
      for (int dt = 0; dt < 8; ++dt) {
        bf16x16 bv = load_frag_b(ldsV, dt * 16, 128, kb * 32);
        o[dt] = wmma_bf16(ap, bv, o[dt]);
      }
    }
  }

  float inv[8];
#pragma unroll
  for (int r = 0; r < 8; ++r) inv[r] = 1.0f / lrow[r];
#pragma unroll
  for (int dt = 0; dt < 8; ++dt) {
#pragma unroll
    for (int r = 0; r < 8; ++r) {
      int t = bb * SEQ + q0 + r + 8 * hi;
      int f = h * HDIM + dt * 16 + n;
      Oat[(size_t)t * HIDDEN + f] = f32_to_bf16(o[dt][r] * inv[r]);
    }
  }
}

// ---------------------------------------------------------------------------
// Output projection: attn_bf16 [4096,2048] x Wo^T -> fp32 out, same 64x64
// wave tiling as the QKV GEMM.
// ---------------------------------------------------------------------------
__global__ __launch_bounds__(256) void out_gemm_kernel(
    const unsigned short* __restrict__ X,
    const unsigned short* __restrict__ Wo,
    float* __restrict__ out) {
  int w = threadIdx.x >> 5, lane = threadIdx.x & 31;
  int m0 = blockIdx.x * 128 + (w >> 2) * 64;
  int n0 = blockIdx.y * 256 + (w & 3) * 64;
  v8f acc[4][4] = {};
  for (int k0 = 0; k0 < HIDDEN; k0 += 32) {
    bf16x16 a[4], b[4];
#pragma unroll
    for (int i = 0; i < 4; ++i) a[i] = load_frag_a(X, m0 + i * 16, HIDDEN, k0);
#pragma unroll
    for (int j = 0; j < 4; ++j) b[j] = load_frag_b(Wo, n0 + j * 16, HIDDEN, k0);
#pragma unroll
    for (int i = 0; i < 4; ++i)
#pragma unroll
      for (int j = 0; j < 4; ++j) acc[i][j] = wmma_bf16(a[i], b[j], acc[i][j]);
  }
  int n = lane & 15, hi = lane >> 4;
#pragma unroll
  for (int i = 0; i < 4; ++i)
#pragma unroll
    for (int j = 0; j < 4; ++j)
#pragma unroll
      for (int r = 0; r < 8; ++r)
        out[(size_t)(m0 + i * 16 + r + 8 * hi) * HIDDEN +
            (n0 + j * 16 + n)] = acc[i][j][r];
}

// ---------------------------------------------------------------------------
extern "C" void kernel_launch(void* const* d_in, const int* in_sizes, int n_in,
                              void* d_out, int out_size, void* d_ws,
                              size_t ws_size, hipStream_t stream) {
  (void)in_sizes; (void)n_in; (void)out_size; (void)ws_size;
  const float* hs = (const float*)d_in[0];
  const float* Wq = (const float*)d_in[1];
  const float* Wk = (const float*)d_in[2];
  const float* Wv = (const float*)d_in[3];
  const float* Wo = (const float*)d_in[4];
  float* out = (float*)d_out;

  // workspace layout (bytes); total ~117.5 MB
  char* ws = (char*)d_ws;
  unsigned short* hsb  = (unsigned short*)(ws + 0);          // 16 MB
  unsigned short* wqb  = (unsigned short*)(ws + 16777216);   // 8 MB each
  unsigned short* wkb  = (unsigned short*)(ws + 25165824);
  unsigned short* wvb  = (unsigned short*)(ws + 33554432);
  unsigned short* wob  = (unsigned short*)(ws + 41943040);
  unsigned short* Qb   = (unsigned short*)(ws + 50331648);   // [B*H,S,D]
  unsigned short* Kb   = (unsigned short*)(ws + 67108864);   // [B*H,S,D]
  unsigned short* Vtb  = (unsigned short*)(ws + 83886080);   // [B*H,D,S]
  unsigned short* attb = (unsigned short*)(ws + 100663296);  // [T,E]

  const int nHS = TOKENS * HIDDEN;   // 8,388,608
  const int nW  = HIDDEN * HIDDEN;   // 4,194,304 per weight
  cvt_f32_bf16_kernel<<<(nHS + 255) / 256, 256, 0, stream>>>(hs, hsb, nHS);
  cvt_f32_bf16_kernel<<<(nW + 255) / 256, 256, 0, stream>>>(Wq, wqb, nW);
  cvt_f32_bf16_kernel<<<(nW + 255) / 256, 256, 0, stream>>>(Wk, wkb, nW);
  cvt_f32_bf16_kernel<<<(nW + 255) / 256, 256, 0, stream>>>(Wv, wvb, nW);
  cvt_f32_bf16_kernel<<<(nW + 255) / 256, 256, 0, stream>>>(Wo, wob, nW);

  qkv_gemm_kernel<<<dim3(TOKENS / 128, HIDDEN / 256, 3), 256, 0, stream>>>(
      hsb, wqb, wkb, wvb, Qb, Kb, Vtb);

  attn_kernel<<<dim3(SEQ / 128, BATCH * NHEADS), 256, 96 * 1024, stream>>>(
      Qb, Kb, Vtb, attb);

  out_gemm_kernel<<<dim3(TOKENS / 128, HIDDEN / 256), 256, 0, stream>>>(
      attb, wob, out);
}
